// AxialCrossMambaUni_89008902242481
// MI455X (gfx1250) — matrix operand order repaired and checked by reading, not compiled, not run on hardware
//
#include <hip/hip_runtime.h>
#include <hip/hip_bf16.h>
#include <cstdint>
#include <cstddef>

typedef __attribute__((ext_vector_type(16))) _Float16 v16h;
typedef __attribute__((ext_vector_type(8)))  _Float16 v8h;
typedef __attribute__((ext_vector_type(8)))  float    v8f;

namespace {
constexpr int kB = 4, kC = 192, kH = 64, kW = 64, kL = kH * kW;   // 4096
constexpr int kDS = 16, kDC = 4, kDI = 384, kDTR = 12;
constexpr int kNdbc = 44, kNdbcP = 48;        // dbc width, padded to 3x16
constexpr int kM = kB * kL;                   // 16384 rows per direction
constexpr int kNin = 2 * kDI;                 // 768
constexpr int kNCH = 32;                      // scan chunks per (dir,b)
constexpr int kCL = kL / kNCH;                // 128 steps per chunk
}

// ---------------- CDNA5 WMMA fragment loaders (f16 16x16x32) ----------------
// A (16x32, MxK): lanes 0-15 row M=lane,   K halves = {k0..k0+7, k0+16..k0+23}
//                 lanes16-31 row M=lane-16,K halves = {k0+8..k0+15, k0+24..k0+31}
__device__ __forceinline__ v16h frag_a(const _Float16* row_base, int k0, int lane) {
  int sel = (lane >> 4) & 1;
  const _Float16* p = row_base + k0 + sel * 8;
  v8h lo = *(const v8h*)(p);
  v8h hi = *(const v8h*)(p + 16);
  v16h f;
#pragma unroll
  for (int i = 0; i < 8; ++i) { f[i] = lo[i]; f[i + 8] = hi[i]; }
  return f;
}

// B (32x16, KxN) stored column-major Bt[n][K]:
// lanes 0-15: col n=lane,    K halves = {k0..k0+7, k0+16..k0+23}
// lanes16-31: col n=lane-16, K halves = {k0+8..k0+15, k0+24..k0+31}
__device__ __forceinline__ v16h frag_b(const _Float16* col_base, int k0, int lane) {
  int sel = (lane >> 4) & 1;
  const _Float16* p = col_base + k0 + sel * 8;
  v8h lo = *(const v8h*)(p);
  v8h hi = *(const v8h*)(p + 16);
  v16h f;
#pragma unroll
  for (int i = 0; i < 8; ++i) { f[i] = lo[i]; f[i + 8] = hi[i]; }
  return f;
}

// One wave computes a 16 x (16*NT) strip; K fully unrolled at compile time.
template<int K, int NT>
__device__ __forceinline__ void wmma_tiles(const _Float16* A, const _Float16* Bt,
                                           int row0, int n0, int lane, v8f acc[NT]) {
  const _Float16* arow = A + (size_t)(row0 + (lane & 15)) * K;
  const _Float16* bcol[NT];
#pragma unroll
  for (int j = 0; j < NT; ++j)
    bcol[j] = Bt + (size_t)(n0 + j * 16 + (lane & 15)) * K;
#pragma unroll
  for (int k0 = 0; k0 < K; k0 += 32) {
    v16h a = frag_a(arow, k0, lane);
#pragma unroll
    for (int j = 0; j < NT; ++j) {
      v16h b = frag_b(bcol[j], k0, lane);
      acc[j] = __builtin_amdgcn_wmma_f32_16x16x32_f16(false, a, false, b,
                                                      (short)0, acc[j], false, false);
    }
  }
}

// ---------------- weight prep (transpose to N-major f16) ----------------
__global__ void k_prep_win(const float* __restrict__ W_in, _Float16* __restrict__ Wint) {
  int id = blockIdx.x * blockDim.x + threadIdx.x;
  if (id >= 4 * kNin * kC) return;
  int k = id % kC, n = (id / kC) % kNin, dir = id / (kC * kNin);
  Wint[((size_t)dir * kNin + n) * kC + k] =
      (_Float16)W_in[((size_t)dir * kC + k) * kNin + n];
}
__global__ void k_prep_wx(const float* __restrict__ W_x, _Float16* __restrict__ Wxt) {
  int id = blockIdx.x * blockDim.x + threadIdx.x;
  if (id >= 4 * kNdbcP * kDI) return;
  int k = id % kDI, n = (id / kDI) % kNdbcP, dir = id / (kDI * kNdbcP);
  float v = (n < kNdbc) ? W_x[((size_t)dir * kDI + k) * kNdbc + n] : 0.f;
  Wxt[((size_t)dir * kNdbcP + n) * kDI + k] = (_Float16)v;
}
__global__ void k_prep_wout(const float* __restrict__ W_out, _Float16* __restrict__ Woutt) {
  int id = blockIdx.x * blockDim.x + threadIdx.x;
  if (id >= 4 * kC * kDI) return;
  int k = id % kDI, n = (id / kDI) % kC, dir = id / (kDI * kC);
  Woutt[((size_t)dir * kC + n) * kDI + k] =
      (_Float16)W_out[((size_t)dir * kDI + k) * kC + n];
}

// ---------------- gather the 4 directional token orders ----------------
__global__ void k_tok(const float* __restrict__ x, const int* __restrict__ idx,
                      _Float16* __restrict__ tok) {
  int id = blockIdx.x * blockDim.x + threadIdx.x;
  if (id >= 4 * kB * kL * kC) return;
  int c = id % kC;
  int t = (id / kC) % kL;
  int b = (id / (kC * kL)) % kB;
  int dir = id / (kC * kL * kB);
  int p;
  if (dir == 0)      p = t;                              // row order
  else if (dir == 1) p = (t & 63) * kW + (t >> 6);       // col order: t = w*H + h
  else {
    int q = idx[t];                                      // diagonal gather
    p = (dir == 2) ? q : (q - (q & 63)) + (63 - (q & 63)); // anti: flip within row
  }
  float v = x[((size_t)b * kC + c) * kL + p];
  tok[((size_t)(dir * kB + b) * kL + t) * kC + c] = (_Float16)v;
}

// ---------------- GEMM1: tok @ W_in -> xi (f16) and silu(z) (f16) ----------------
__global__ void k_gemm_in(const _Float16* __restrict__ tok, const _Float16* __restrict__ Wint,
                          _Float16* __restrict__ xi, _Float16* __restrict__ zs) {
  constexpr int NT = 4, NG = kNin / (16 * NT);   // 12 n-groups of 64
  constexpr int MT = kM / 16;                    // 1024 m-tiles
  int wave = (blockIdx.x * blockDim.x + threadIdx.x) >> 5;
  int lane = threadIdx.x & 31;
  int per_dir = MT * NG;
  if (wave >= 4 * per_dir) return;
  int dir = wave / per_dir, rem = wave - dir * per_dir;
  int mt = rem / NG, ng = rem - mt * NG;
  int row0 = mt * 16, n0 = ng * (16 * NT);
  const _Float16* A  = tok  + (size_t)dir * kM * kC;
  const _Float16* Bt = Wint + (size_t)dir * kNin * kC;
  v8f acc[NT];
#pragma unroll
  for (int j = 0; j < NT; ++j)
#pragma unroll
    for (int r = 0; r < 8; ++r) acc[j][r] = 0.f;
  wmma_tiles<kC, NT>(A, Bt, row0, n0, lane, acc);
  int sel = (lane >> 4) & 1, nl = lane & 15;
#pragma unroll
  for (int j = 0; j < NT; ++j)
#pragma unroll
    for (int r = 0; r < 8; ++r) {
      int row = row0 + r + 8 * sel;
      int col = n0 + j * 16 + nl;
      float v = acc[j][r];
      if (col < kDI) {
        xi[((size_t)dir * kM + row) * kDI + col] = (_Float16)v;
      } else {                                   // z half: fuse silu
        float s = v / (1.f + __expf(-v));
        zs[((size_t)dir * kM + row) * kDI + (col - kDI)] = (_Float16)s;
      }
    }
}

// ---------------- depthwise causal conv (4-tap) + silu ----------------
__global__ void k_conv(const _Float16* __restrict__ xi, const float* __restrict__ conv_w,
                       const float* __restrict__ conv_b, _Float16* __restrict__ xs) {
  int id = blockIdx.x * blockDim.x + threadIdx.x;
  if (id >= 4 * kB * kL * kDI) return;
  int c = id % kDI;
  int t = (id / kDI) % kL;
  int pair = id / (kDI * kL);                    // dir*4 + b
  int dir = pair >> 2;
  float acc = conv_b[dir * kDI + c];
  const float* w = conv_w + ((size_t)dir * kDI + c) * kDC;
#pragma unroll
  for (int k = 0; k < kDC; ++k) {
    int tt = t - (kDC - 1) + k;
    if (tt >= 0)
      acc += w[k] * (float)xi[((size_t)pair * kL + tt) * kDI + c];
  }
  acc = acc / (1.f + __expf(-acc));              // silu
  xs[((size_t)pair * kL + t) * kDI + c] = (_Float16)acc;
}

// ---------------- GEMM2: xs @ W_x -> dbc (f32, padded to 48) ----------------
__global__ void k_gemm_x(const _Float16* __restrict__ xs, const _Float16* __restrict__ Wxt,
                         float* __restrict__ dbc) {
  constexpr int NT = 3;                          // 48 cols in one group
  constexpr int MT = kM / 16;
  int wave = (blockIdx.x * blockDim.x + threadIdx.x) >> 5;
  int lane = threadIdx.x & 31;
  if (wave >= 4 * MT) return;
  int dir = wave / MT, mt = wave - dir * MT;
  int row0 = mt * 16;
  const _Float16* A  = xs  + (size_t)dir * kM * kDI;
  const _Float16* Bt = Wxt + (size_t)dir * kNdbcP * kDI;
  v8f acc[NT];
#pragma unroll
  for (int j = 0; j < NT; ++j)
#pragma unroll
    for (int r = 0; r < 8; ++r) acc[j][r] = 0.f;
  wmma_tiles<kDI, NT>(A, Bt, row0, 0, lane, acc);
  int sel = (lane >> 4) & 1, nl = lane & 15;
#pragma unroll
  for (int j = 0; j < NT; ++j)
#pragma unroll
    for (int r = 0; r < 8; ++r) {
      int row = row0 + r + 8 * sel;
      int col = j * 16 + nl;
      dbc[((size_t)dir * kM + row) * kNdbcP + col] = acc[j][r];
    }
}

// ---------------- blocked selective scan ----------------
// Recurrence h_t = exp(dt_t*A)*h_{t-1} + dt_t*x_t*B_t is associative; the chunk
// decay product collapses to exp(A * sum(dt)). Pass1: chunk-local state + decay.
// Combine: serial over 32 chunks (tiny). Pass2: re-run chunk from true h_in.

// per-step shared helpers: dbc row double-buffered in LDS, next-row load issued
// before compute so L2 latency overlaps the exp chain.

__global__ void __launch_bounds__(kDI) k_scan1(
    const float* __restrict__ dbc, const _Float16* __restrict__ xs,
    const float* __restrict__ W_dt, const float* __restrict__ b_dt,
    const float* __restrict__ A_log, float* __restrict__ hloc,
    float* __restrict__ aprod) {
  int blk = blockIdx.x;
  int pair = blk / kNCH, ch = blk % kNCH;        // pair = dir*4 + b
  int dir = pair >> 2;
  int d = threadIdx.x;
  int tid = threadIdx.x;
  __shared__ float srow[2][kNdbcP];
  float wdt[kDTR];
#pragma unroll
  for (int k = 0; k < kDTR; ++k) wdt[k] = W_dt[((size_t)dir * kDTR + k) * kDI + d];
  float bdt = b_dt[dir * kDI + d];
  float Aa[kDS];
#pragma unroll
  for (int s = 0; s < kDS; ++s)
    Aa[s] = -__expf(A_log[((size_t)dir * kDI + d) * kDS + s]);
  float h[kDS];
#pragma unroll
  for (int s = 0; s < kDS; ++s) h[s] = 0.f;
  float dtsum = 0.f;

  size_t rowbase = (size_t)pair * kL + (size_t)ch * kCL;
  if (tid < kNdbcP) srow[0][tid] = dbc[rowbase * kNdbcP + tid];
  float xv = (float)xs[rowbase * kDI + d];
  __syncthreads();

  for (int tt = 0; tt < kCL; ++tt) {
    int cur = tt & 1, nxt = cur ^ 1;
    float nv = 0.f, xn = 0.f;
    if (tt + 1 < kCL) {
      if (tid < kNdbcP) nv = dbc[(rowbase + tt + 1) * kNdbcP + tid];
      xn = (float)xs[(rowbase + tt + 1) * kDI + d];
    }
    if (tt + 6 < kCL) {
      if (tid < kNdbcP) __builtin_prefetch(&dbc[(rowbase + tt + 6) * kNdbcP + tid], 0, 1);
      __builtin_prefetch(&xs[(rowbase + tt + 6) * kDI + d], 0, 1);
    }
    float dtr = bdt;
#pragma unroll
    for (int k = 0; k < kDTR; ++k) dtr += srow[cur][k] * wdt[k];
    float dt = (dtr > 20.f) ? dtr : __logf(1.f + __expf(dtr));   // softplus
    dtsum += dt;
    float dx = dt * xv;
#pragma unroll
    for (int s = 0; s < kDS; ++s)
      h[s] = h[s] * __expf(dt * Aa[s]) + dx * srow[cur][kDTR + s]; // B at [12..27]
    __syncthreads();
    if (tt + 1 < kCL && tid < kNdbcP) srow[nxt][tid] = nv;
    xv = xn;
    __syncthreads();
  }
  size_t obase = (((size_t)blk) * kDI + d) * kDS;
#pragma unroll
  for (int s = 0; s < kDS; ++s) {
    hloc[obase + s]  = h[s];
    aprod[obase + s] = __expf(Aa[s] * dtsum);    // chunk decay product
  }
}

// serial chunk combine; rewrites aprod[blk] in place with the chunk's INPUT state
__global__ void __launch_bounds__(kDI) k_comb(const float* __restrict__ hloc,
                                              float* __restrict__ aprod) {
  int pair = blockIdx.x;
  int d = threadIdx.x;
  float carry[kDS];
#pragma unroll
  for (int s = 0; s < kDS; ++s) carry[s] = 0.f;
  for (int ch = 0; ch < kNCH; ++ch) {
    size_t base = (((size_t)pair * kNCH + ch) * kDI + d) * kDS;
#pragma unroll
    for (int s = 0; s < kDS; ++s) {
      float a  = aprod[base + s];
      float hl = hloc[base + s];
      aprod[base + s] = carry[s];                // h_in for chunk ch
      carry[s] = a * carry[s] + hl;
    }
  }
}

__global__ void __launch_bounds__(kDI) k_scan2(
    const float* __restrict__ dbc, const _Float16* __restrict__ xs,
    const _Float16* __restrict__ zs, const float* __restrict__ W_dt,
    const float* __restrict__ b_dt, const float* __restrict__ A_log,
    const float* __restrict__ D_skip, const float* __restrict__ hin,
    _Float16* __restrict__ yg) {
  int blk = blockIdx.x;
  int pair = blk / kNCH, ch = blk % kNCH;
  int dir = pair >> 2;
  int d = threadIdx.x;
  int tid = threadIdx.x;
  __shared__ float srow[2][kNdbcP];
  float wdt[kDTR];
#pragma unroll
  for (int k = 0; k < kDTR; ++k) wdt[k] = W_dt[((size_t)dir * kDTR + k) * kDI + d];
  float bdt = b_dt[dir * kDI + d];
  float Aa[kDS];
#pragma unroll
  for (int s = 0; s < kDS; ++s)
    Aa[s] = -__expf(A_log[((size_t)dir * kDI + d) * kDS + s]);
  float dsk = D_skip[dir * kDI + d];
  float h[kDS];
  size_t ibase = (((size_t)blk) * kDI + d) * kDS;
#pragma unroll
  for (int s = 0; s < kDS; ++s) h[s] = hin[ibase + s];   // true chunk input state

  size_t rowbase = (size_t)pair * kL + (size_t)ch * kCL;
  if (tid < kNdbcP) srow[0][tid] = dbc[rowbase * kNdbcP + tid];
  float xv = (float)xs[rowbase * kDI + d];
  float zv = (float)zs[rowbase * kDI + d];
  __syncthreads();

  for (int tt = 0; tt < kCL; ++tt) {
    int cur = tt & 1, nxt = cur ^ 1;
    float nv = 0.f, xn = 0.f, zn = 0.f;
    if (tt + 1 < kCL) {
      if (tid < kNdbcP) nv = dbc[(rowbase + tt + 1) * kNdbcP + tid];
      xn = (float)xs[(rowbase + tt + 1) * kDI + d];
      zn = (float)zs[(rowbase + tt + 1) * kDI + d];
    }
    if (tt + 6 < kCL) {
      if (tid < kNdbcP) __builtin_prefetch(&dbc[(rowbase + tt + 6) * kNdbcP + tid], 0, 1);
      __builtin_prefetch(&xs[(rowbase + tt + 6) * kDI + d], 0, 1);
      __builtin_prefetch(&zs[(rowbase + tt + 6) * kDI + d], 0, 1);
    }
    float dtr = bdt;
#pragma unroll
    for (int k = 0; k < kDTR; ++k) dtr += srow[cur][k] * wdt[k];
    float dt = (dtr > 20.f) ? dtr : __logf(1.f + __expf(dtr));
    float dx = dt * xv;
    float y = 0.f;
#pragma unroll
    for (int s = 0; s < kDS; ++s) {
      h[s] = h[s] * __expf(dt * Aa[s]) + dx * srow[cur][kDTR + s];
      y += h[s] * srow[cur][kDTR + kDS + s];     // C at [28..43]
    }
    y += xv * dsk;
    yg[(rowbase + tt) * kDI + d] = (_Float16)(y * zv);
    __syncthreads();
    if (tt + 1 < kCL && tid < kNdbcP) srow[nxt][tid] = nv;
    xv = xn; zv = zn;
    __syncthreads();
  }
}

// ---------------- GEMM3: yg @ W_out -> per-direction output (f16) ----------------
__global__ void k_gemm_out(const _Float16* __restrict__ yg, const _Float16* __restrict__ Woutt,
                           _Float16* __restrict__ dout) {
  constexpr int NT = 4, NG = kC / (16 * NT);     // 3 n-groups of 64
  constexpr int MT = kM / 16;
  int wave = (blockIdx.x * blockDim.x + threadIdx.x) >> 5;
  int lane = threadIdx.x & 31;
  int per_dir = MT * NG;
  if (wave >= 4 * per_dir) return;
  int dir = wave / per_dir, rem = wave - dir * per_dir;
  int mt = rem / NG, ng = rem - mt * NG;
  int row0 = mt * 16, n0 = ng * (16 * NT);
  const _Float16* A  = yg    + (size_t)dir * kM * kDI;
  const _Float16* Bt = Woutt + (size_t)dir * kC * kDI;
  v8f acc[NT];
#pragma unroll
  for (int j = 0; j < NT; ++j)
#pragma unroll
    for (int r = 0; r < 8; ++r) acc[j][r] = 0.f;
  wmma_tiles<kDI, NT>(A, Bt, row0, n0, lane, acc);
  int sel = (lane >> 4) & 1, nl = lane & 15;
#pragma unroll
  for (int j = 0; j < NT; ++j)
#pragma unroll
    for (int r = 0; r < 8; ++r) {
      int row = row0 + r + 8 * sel;
      int col = n0 + j * 16 + nl;
      dout[((size_t)dir * kM + row) * kC + col] = (_Float16)acc[j][r];
    }
}

// ---------------- final gate: inverse-permute 4 directions, sigmoid, multiply x ----------------
__global__ void k_gate(const float* __restrict__ x, const _Float16* __restrict__ dout,
                       const int* __restrict__ inv_idx, float* __restrict__ out) {
  int id = blockIdx.x * blockDim.x + threadIdx.x;
  if (id >= kB * kC * kL) return;
  int p = id % kL;
  int c = (id / kL) % kC;
  int b = id / (kL * kC);
  int i = p >> 6, j = p & 63;
  int t0 = p;
  int t1 = j * kH + i;                            // col order inverse
  int t2 = inv_idx[p];                            // diag inverse
  int t3 = inv_idx[i * kW + (kW - 1 - j)];        // anti-diag inverse (flip)
  size_t base = (size_t)b * kL;
  float s = (float)dout[((size_t)0 * kM + base + t0) * kC + c]
          + (float)dout[((size_t)1 * kM + base + t1) * kC + c]
          + (float)dout[((size_t)2 * kM + base + t2) * kC + c]
          + (float)dout[((size_t)3 * kM + base + t3) * kC + c];
  float g = 1.f / (1.f + __expf(-0.25f * s));
  size_t xo = ((size_t)b * kC + c) * kL + p;
  out[xo] = x[xo] * g;
}

// ---------------- host launch ----------------
static inline int ceil_div(long long a, int b) { return (int)((a + b - 1) / b); }

extern "C" void kernel_launch(void* const* d_in, const int* in_sizes, int n_in,
                              void* d_out, int out_size, void* d_ws, size_t ws_size,
                              hipStream_t stream) {
  const float* x      = (const float*)d_in[0];
  const float* W_in   = (const float*)d_in[1];
  const float* conv_w = (const float*)d_in[2];
  const float* conv_b = (const float*)d_in[3];
  const float* W_x    = (const float*)d_in[4];
  const float* W_dt   = (const float*)d_in[5];
  const float* b_dt   = (const float*)d_in[6];
  const float* A_log  = (const float*)d_in[7];
  const float* D_skip = (const float*)d_in[8];
  const float* W_out  = (const float*)d_in[9];
  const int*   idx    = (const int*)d_in[10];
  const int*   invix  = (const int*)d_in[11];
  float* out = (float*)d_out;

  char* ws = (char*)d_ws;
  size_t off = 0;
  auto carve = [&](size_t bytes) { char* p = ws + off; off += (bytes + 255) & ~(size_t)255; return p; };
  _Float16* tok   = (_Float16*)carve((size_t)4 * kM * kC      * 2);
  _Float16* Wint  = (_Float16*)carve((size_t)4 * kNin * kC    * 2);
  _Float16* Wxt   = (_Float16*)carve((size_t)4 * kNdbcP * kDI * 2);
  _Float16* Woutt = (_Float16*)carve((size_t)4 * kC * kDI     * 2);
  _Float16* xi    = (_Float16*)carve((size_t)4 * kM * kDI     * 2);
  _Float16* zs    = (_Float16*)carve((size_t)4 * kM * kDI     * 2);
  _Float16* xs    = (_Float16*)carve((size_t)4 * kM * kDI     * 2);
  _Float16* yg    = (_Float16*)carve((size_t)4 * kM * kDI     * 2);
  float*    dbc   = (float*)   carve((size_t)4 * kM * kNdbcP  * 4);
  _Float16* dout  = (_Float16*)carve((size_t)4 * kM * kC      * 2);
  float*    hloc  = (float*)   carve((size_t)16 * kNCH * kDI * kDS * 4);
  float*    aprod = (float*)   carve((size_t)16 * kNCH * kDI * kDS * 4); // reused as h_in
  (void)ws_size; (void)in_sizes; (void)n_in; (void)out_size;

  const int T = 256;
  k_prep_win <<<ceil_div((long long)4 * kNin * kC, T), T, 0, stream>>>(W_in, Wint);
  k_prep_wx  <<<ceil_div((long long)4 * kNdbcP * kDI, T), T, 0, stream>>>(W_x, Wxt);
  k_prep_wout<<<ceil_div((long long)4 * kC * kDI, T), T, 0, stream>>>(W_out, Woutt);
  k_tok      <<<ceil_div((long long)4 * kB * kL * kC, T), T, 0, stream>>>(x, idx, tok);

  // GEMM1: 4 dirs * 1024 m-tiles * 12 n-groups = 49152 waves
  k_gemm_in  <<<ceil_div((long long)4 * (kM / 16) * 12 * 32, T), T, 0, stream>>>(tok, Wint, xi, zs);
  k_conv     <<<ceil_div((long long)4 * kB * kL * kDI, T), T, 0, stream>>>(xi, conv_w, conv_b, xs);
  // GEMM2: 4 * 1024 waves
  k_gemm_x   <<<ceil_div((long long)4 * (kM / 16) * 32, T), T, 0, stream>>>(xs, Wxt, dbc);

  // blocked scan: 512 blocks pass1/pass2, 16-block serial combine
  k_scan1    <<<dim3(16 * kNCH), dim3(kDI), 0, stream>>>(dbc, xs, W_dt, b_dt, A_log, hloc, aprod);
  k_comb     <<<dim3(16), dim3(kDI), 0, stream>>>(hloc, aprod);
  k_scan2    <<<dim3(16 * kNCH), dim3(kDI), 0, stream>>>(dbc, xs, zs, W_dt, b_dt, A_log,
                                                         D_skip, aprod, yg);

  // GEMM3: 4 * 1024 * 3 waves
  k_gemm_out <<<ceil_div((long long)4 * (kM / 16) * 3 * 32, T), T, 0, stream>>>(yg, Woutt, dout);
  k_gate     <<<ceil_div((long long)kB * kC * kL, T), T, 0, stream>>>(x, dout, invix, out);
}
